// IPA_25168508354959
// MI455X (gfx1250) — matrix operand order repaired and checked by reading, not compiled, not run on hardware
//
#include <hip/hip_runtime.h>
#include <hip/hip_bf16.h>

// ---------------------------------------------------------------------------
// CDNA5 (gfx1250) implementation of the AlphaFold structure-module reference.
// All matmuls run through v_wmma_f32_16x16x32_bf16 (bf16 operands, f32 acc).
// z is layernormed ONCE into bf16 (67MB -> resident in the 192MB L2) and the
// pair-bias projection + a0 are hoisted out of the 8-block loop.
// LDS tiles are staged in *fragment order* so each WMMA operand is fed by two
// aligned ds_load_b128 per matrix instead of 16 ds_load_u16 gathers.
// ---------------------------------------------------------------------------

typedef __bf16 bf16;
typedef __attribute__((ext_vector_type(16))) __bf16 v16bf;
typedef __attribute__((ext_vector_type(8)))  __bf16 v8bf;
typedef __attribute__((ext_vector_type(8)))  float  v8f;

#define CDIV(a,b) (((a)+(b)-1)/(b))

static constexpr int NRES = 512;
static constexpr int CS   = 384;
static constexpr int CZ   = 128;
static constexpr int NH   = 12;
static constexpr int CHID = 16;
static constexpr int PQ   = 4;
static constexpr int PV   = 8;

__device__ __forceinline__ unsigned short f2bf_u16(float f) {
  unsigned u = __builtin_bit_cast(unsigned, f);
  u += 0x7FFFu + ((u >> 16) & 1u);             // round-to-nearest-even
  return (unsigned short)(u >> 16);
}
__device__ __forceinline__ bf16 f2bf(float f) {
  unsigned short h = f2bf_u16(f);
  return __builtin_bit_cast(bf16, h);
}
__device__ __forceinline__ float bf2f(bf16 b) {
  unsigned short h = __builtin_bit_cast(unsigned short, b);
  unsigned u = ((unsigned)h) << 16;
  return __builtin_bit_cast(float, u);
}
__device__ __forceinline__ float cvt_in(float v) { return v; }
__device__ __forceinline__ float cvt_in(bf16 v)  { return bf2f(v); }
__device__ __forceinline__ void st_o(float* p, float v) { *p = v; }
__device__ __forceinline__ void st_o(bf16* p, float v)  { *p = f2bf(v); }

// ---------------------------------------------------------------------------
// Batched, strided GEMM:  C[b,m,n] = act( sum_k A[b,m,k]*B[b,k,n] + bias[n]
//                                         + resid[b,m,n] )
// Workgroup = WM*WN waves (8 -> 256 threads), each wave one 16x16 WMMA tile.
//
// LDS is laid out in WMMA *fragment order* (CDNA5 ISA 7.12.2, wave32):
//   A (16x32 bf16): fragment lane = (row&15)+16*half, elements e=0..15 where
//        k = (e<8 ? e : 8+e) ... inverted here: kk -> half=(kk>>3)&1,
//        base=kk-8*half, e = base<8 ? base : base-8
//   B (32x16 bf16): fragment lane = (col&15)+16*((kk>>3)&1), e = (kk&7)+8*(kk>>4)
//   C/D (16x16 f32): col = lane&15, row = r + 8*(lane>>4)
// Each lane's 16 bf16 fragment elements are contiguous (32B, 16B-aligned),
// loaded as two ds_load_b128. 48B per-lane stride breaks bank-collision period.
// ---------------------------------------------------------------------------
template <int WM, int WN, typename TA, typename TB>
__global__ __launch_bounds__(256)
void gemm_wmma(const TA* __restrict__ A, long long a_bs, long long lda_m, long long lda_k, int relu_a,
               const TB* __restrict__ Bm, long long b_bs, long long ldb_k, long long ldb_n,
               const float* __restrict__ bias, const float* resid,
               float* C, long long c_bs, long long ldc_m, long long ldc_n,
               int M, int Nc, int K, int relu_out)
{
  constexpr int TM = WM * 16, TN = WN * 16, T = 32 * WM * WN;
  __shared__ alignas(16) unsigned short Asf[WM][32][24];   // [subtile][frag lane][16 elems + pad]
  __shared__ alignas(16) unsigned short Bsf[WN][32][24];

  const int tid  = threadIdx.x;
  const int wave = tid >> 5, lane = tid & 31;
  const int wm = wave / WN, wn = wave % WN;
  const int m0 = blockIdx.x * TM;
  const int n0 = blockIdx.y * TN;
  const long long batch = blockIdx.z;
  const TA* Ab = A  + batch * a_bs;
  const TB* Bb = Bm + batch * b_bs;

  v8f acc = {0.f,0.f,0.f,0.f,0.f,0.f,0.f,0.f};

  for (int k0 = 0; k0 < K; k0 += 32) {
    // one global_prefetch_b8 per thread for the next A K-step
    if (k0 + 32 < K) {
      int pm = m0 + (tid % TM);
      if (pm < M)
        __builtin_prefetch(&Ab[(long long)pm * lda_m + (long long)(k0 + 32) * lda_k], 0, 1);
    }

    // ---- stage A tile (TM x 32) in fragment order; 2 k's -> one b32 store ----
    for (int idx = tid; idx < TM * 16; idx += T) {
      int r = idx >> 4, q = idx & 15;           // q = kk pair index
      int kk0 = 2 * q;
      int half = (kk0 >> 3) & 1;
      int base = kk0 - 8 * half;
      int e0   = (base < 8) ? base : base - 8;  // even fragment element
      int flane = (r & 15) + 16 * half;
      int m = m0 + r;
      float v0 = 0.f, v1 = 0.f;
      if (m < M) {
        const TA* ap = &Ab[(long long)m * lda_m + (long long)(k0 + kk0) * lda_k];
        if (k0 + kk0     < K) v0 = cvt_in(ap[0]);
        if (k0 + kk0 + 1 < K) v1 = cvt_in(ap[lda_k]);
        if (relu_a) { v0 = fmaxf(v0, 0.f); v1 = fmaxf(v1, 0.f); }
      }
      unsigned pk = (unsigned)f2bf_u16(v0) | ((unsigned)f2bf_u16(v1) << 16);
      *(unsigned*)&Asf[r >> 4][flane][e0] = pk;
    }
    // ---- stage B tile (32 x TN) in fragment order ----
    for (int idx = tid; idx < 16 * TN; idx += T) {
      int q = idx / TN, n = idx % TN;           // q = kk pair index
      int kk0 = 2 * q;
      int e0  = (kk0 & 7) + 8 * (kk0 >> 4);
      int flane = (n & 15) + 16 * ((kk0 >> 3) & 1);
      int nn = n0 + n;
      float v0 = 0.f, v1 = 0.f;
      if (nn < Nc) {
        const TB* bp = &Bb[(long long)(k0 + kk0) * ldb_k + (long long)nn * ldb_n];
        if (k0 + kk0     < K) v0 = cvt_in(bp[0]);
        if (k0 + kk0 + 1 < K) v1 = cvt_in(bp[ldb_k]);
      }
      unsigned pk = (unsigned)f2bf_u16(v0) | ((unsigned)f2bf_u16(v1) << 16);
      *(unsigned*)&Bsf[n >> 4][flane][e0] = pk;
    }
    __syncthreads();

    // ---- fragments: 2x ds_load_b128 each, then WMMA ----
    const unsigned short* ap = &Asf[wm][lane][0];
    const unsigned short* bp = &Bsf[wn][lane][0];
    v8bf alo = *(const v8bf*)(ap);
    v8bf ahi = *(const v8bf*)(ap + 8);
    v8bf blo = *(const v8bf*)(bp);
    v8bf bhi = *(const v8bf*)(bp + 8);
    v16bf af  = __builtin_shufflevector(alo, ahi, 0,1,2,3,4,5,6,7,8,9,10,11,12,13,14,15);
    v16bf bfr = __builtin_shufflevector(blo, bhi, 0,1,2,3,4,5,6,7,8,9,10,11,12,13,14,15);
    acc = __builtin_amdgcn_wmma_f32_16x16x32_bf16(false, af, false, bfr,
                                                  (short)0, acc, false, false);
    __syncthreads();
  }

  // ---- epilogue ----
  const int col = lane & 15, half = lane >> 4;
#pragma unroll
  for (int r = 0; r < 8; ++r) {
    int m = m0 + wm * 16 + r + 8 * half;
    int n = n0 + wn * 16 + col;
    if (m < M && n < Nc) {
      float v = acc[r];
      if (bias)  v += bias[n];
      long long off = batch * c_bs + (long long)m * ldc_m + (long long)n * ldc_n;
      if (resid) v += resid[off];
      if (relu_out) v = fmaxf(v, 0.f);
      C[off] = v;
    }
  }
}

// ---------------------------------------------------------------------------
// LayerNorm (one row per block, 128 threads). In-place safe. TO = float|bf16.
// ---------------------------------------------------------------------------
template <typename TO>
__global__ __launch_bounds__(128)
void ln_kernel(const float* x, const float* g, const float* b, TO* out, int C)
{
  long long row = blockIdx.x;
  const float* xr = x + row * C;
  __shared__ float red[128];
  float s = 0.f;
  for (int c = threadIdx.x; c < C; c += 128) s += xr[c];
  red[threadIdx.x] = s; __syncthreads();
  for (int o = 64; o > 0; o >>= 1) { if (threadIdx.x < o) red[threadIdx.x] += red[threadIdx.x + o]; __syncthreads(); }
  float mean = red[0] / C; __syncthreads();
  float vs = 0.f;
  for (int c = threadIdx.x; c < C; c += 128) { float d = xr[c] - mean; vs += d * d; }
  red[threadIdx.x] = vs; __syncthreads();
  for (int o = 64; o > 0; o >>= 1) { if (threadIdx.x < o) red[threadIdx.x] += red[threadIdx.x + o]; __syncthreads(); }
  float inv = rsqrtf(red[0] / C + 1e-5f); __syncthreads();
  for (int c = threadIdx.x; c < C; c += 128)
    st_o(out + row * C + c, (xr[c] - mean) * inv * g[c] + b[c]);
}

// raw (n, 3*P) coord-major -> rotated+translated points (n, P, 3)
__global__ void to_pts_kernel(const float* raw, const float* rot, const float* trans,
                              float* out, int P)
{
  int n = blockIdx.x, p = threadIdx.x;
  if (p >= P) return;
  long long base = (long long)n * 3 * P;
  float x = raw[base + p], y = raw[base + P + p], z = raw[base + 2 * P + p];
  const float* R = rot + n * 9; const float* t = trans + n * 3;
  float* o = out + ((long long)n * P + p) * 3;
  o[0] = R[0]*x + R[1]*y + R[2]*z + t[0];
  o[1] = R[3]*x + R[4]*y + R[5]*z + t[1];
  o[2] = R[6]*x + R[7]*y + R[8]*z + t[2];
}

// logits (qk dot already in `a`) + pair bias + point attention + mask, softmax over j
__global__ __launch_bounds__(256)
void attn_softmax_kernel(float* a, const float* bpair, const float* qp,
                         const float* kvp, const float* head_w, const float* mask)
{
  int i = blockIdx.x, h = blockIdx.y;
  __shared__ float red[256];
  float hw = log1pf(expf(head_w[h])) * 0.13608276f;   // sqrt(1/(3*(PQ*9/2)))
  float qx[PQ], qy[PQ], qz[PQ];
#pragma unroll
  for (int p = 0; p < PQ; ++p) {
    const float* q = qp + ((long long)i * (NH * PQ) + h * PQ + p) * 3;
    qx[p] = q[0]; qy[p] = q[1]; qz[p] = q[2];
  }
  float mi = mask[i];
  float* arow = a + ((long long)h * NRES + i) * NRES;
  float lg[2], mx = -3.0e38f;
  for (int u = 0; u < 2; ++u) {
    int j = threadIdx.x + u * 256;
    float pt = 0.f;
    const float* kb = kvp + ((long long)j * (NH * (PQ + PV)) + h * (PQ + PV)) * 3;
#pragma unroll
    for (int p = 0; p < PQ; ++p) {
      float dx = qx[p] - kb[3*p], dy = qy[p] - kb[3*p+1], dz = qz[p] - kb[3*p+2];
      pt += dx*dx + dy*dy + dz*dz;
    }
    float l = arow[j] * 0.14433757f                                   // sqrt(1/48)
            + 0.57735027f * bpair[((long long)i * NRES + j) * NH + h] // sqrt(1/3)
            - 0.5f * hw * pt
            + 1e8f * (mi * mask[j] - 1.f);
    lg[u] = l; mx = fmaxf(mx, l);
  }
  red[threadIdx.x] = mx; __syncthreads();
  for (int o = 128; o > 0; o >>= 1) { if (threadIdx.x < o) red[threadIdx.x] = fmaxf(red[threadIdx.x], red[threadIdx.x + o]); __syncthreads(); }
  mx = red[0]; __syncthreads();
  float e0 = expf(lg[0] - mx), e1 = expf(lg[1] - mx);
  red[threadIdx.x] = e0 + e1; __syncthreads();
  for (int o = 128; o > 0; o >>= 1) { if (threadIdx.x < o) red[threadIdx.x] += red[threadIdx.x + o]; __syncthreads(); }
  float inv = 1.f / red[0];
  arow[threadIdx.x]       = e0 * inv;
  arow[threadIdx.x + 256] = e1 * inv;
}

// concat [ o | o_pt.x | o_pt.y | o_pt.z | |o_pt| | o_pair ] with inverse-frame transform
__global__ __launch_bounds__(256)
void feats_kernel(const float* o, const float* o_pt_g, const float* o_pair,
                  const float* rot, const float* trans, float* feats)
{
  int i = blockIdx.x, t = threadIdx.x;
  float* f = feats + (long long)i * 2112;
  for (int c = t; c < 192; c += 256) f[c] = o[(long long)i * 192 + c];
  if (t < 96) {
    const float* pg = o_pt_g + ((long long)i * 96 + t) * 3;
    const float* R = rot + i * 9; const float* tr = trans + i * 3;
    float vx = pg[0] - tr[0], vy = pg[1] - tr[1], vz = pg[2] - tr[2];
    float lx = R[0]*vx + R[3]*vy + R[6]*vz;   // R^T v
    float ly = R[1]*vx + R[4]*vy + R[7]*vz;
    float lz = R[2]*vx + R[5]*vy + R[8]*vz;
    f[192 + t] = lx; f[288 + t] = ly; f[384 + t] = lz;
    f[480 + t] = sqrtf(lx*lx + ly*ly + lz*lz + 1e-8f);
  }
  for (int c = t; c < 1536; c += 256) f[576 + c] = o_pair[(long long)i * 1536 + c];
}

__global__ void init_pose_kernel(float* quat, float* trans, float* rot)
{
  int n = blockIdx.x * blockDim.x + threadIdx.x;
  if (n >= NRES) return;
  quat[n*4] = 1.f; quat[n*4+1] = 0.f; quat[n*4+2] = 0.f; quat[n*4+3] = 0.f;
  trans[n*3] = 0.f; trans[n*3+1] = 0.f; trans[n*3+2] = 0.f;
  float* R = rot + n * 9;
  R[0]=1.f;R[1]=0.f;R[2]=0.f; R[3]=0.f;R[4]=1.f;R[5]=0.f; R[6]=0.f;R[7]=0.f;R[8]=1.f;
}

// quaternion composition, trans update (old rot), new rot, frames output
__global__ void bb_update_kernel(const float* upd, float* quat, float* trans,
                                 float* rot, float* out_frames)
{
  int n = blockIdx.x * blockDim.x + threadIdx.x;
  if (n >= NRES) return;
  const float* u = upd + n * 16;
  float q0 = quat[n*4], q1 = quat[n*4+1], q2 = quat[n*4+2], q3 = quat[n*4+3];
  float b = u[0], c = u[1], d = u[2];
  float nw = q0       - q1*b - q2*c - q3*d;
  float nx = q0*b + q1       + q2*d - q3*c;
  float ny = q0*c - q1*d + q2       + q3*b;
  float nz = q0*d + q1*c - q2*b + q3;
  float inv = rsqrtf(nw*nw + nx*nx + ny*ny + nz*nz);
  nw *= inv; nx *= inv; ny *= inv; nz *= inv;
  const float* R = rot + n * 9;          // old rotation (from quat before update)
  float tx = trans[n*3]   + R[0]*u[3] + R[1]*u[4] + R[2]*u[5];
  float ty = trans[n*3+1] + R[3]*u[3] + R[4]*u[4] + R[5]*u[5];
  float tz = trans[n*3+2] + R[6]*u[3] + R[7]*u[4] + R[8]*u[5];
  quat[n*4] = nw; quat[n*4+1] = nx; quat[n*4+2] = ny; quat[n*4+3] = nz;
  trans[n*3] = tx; trans[n*3+1] = ty; trans[n*3+2] = tz;
  float w = nw, x = nx, y = ny, z = nz;
  float* Rn = rot + n * 9;
  Rn[0]=1-2*(y*y+z*z); Rn[1]=2*(x*y-w*z);   Rn[2]=2*(x*z+w*y);
  Rn[3]=2*(x*y+w*z);   Rn[4]=1-2*(x*x+z*z); Rn[5]=2*(y*z-w*x);
  Rn[6]=2*(x*z-w*y);   Rn[7]=2*(y*z+w*x);   Rn[8]=1-2*(x*x+y*y);
  float* fo = out_frames + (long long)n * 7;
  fo[0]=nw; fo[1]=nx; fo[2]=ny; fo[3]=nz;
  fo[4]=tx*10.f; fo[5]=ty*10.f; fo[6]=tz*10.f;
}

__global__ void angle_out_kernel(const float* un_a, float* out_un, float* out_ang)
{
  int n = blockIdx.x, k = threadIdx.x;
  if (k >= 7) return;
  float s = un_a[(long long)n * 14 + 2*k], c = un_a[(long long)n * 14 + 2*k + 1];
  float nrm = sqrtf(fmaxf(s*s + c*c, 1e-8f));
  out_un[(long long)n*14 + 2*k] = s;      out_un[(long long)n*14 + 2*k + 1] = c;
  out_ang[(long long)n*14 + 2*k] = s/nrm; out_ang[(long long)n*14 + 2*k + 1] = c/nrm;
}

// DEFAULT_FRAMES/GROUP_IDX/ATOM_MASK/LIT_POS are all-zero module constants, so
// exactly: g_r = 0, g_t = bb_trans (broadcast over 8 frames), positions = 0.
__global__ __launch_bounds__(128)
void geom_out_kernel(const float* trans, float* out_sc, float* out_pos)
{
  int n = blockIdx.x, t = threadIdx.x;
  float tv[3] = { trans[n*3]*10.f, trans[n*3+1]*10.f, trans[n*3+2]*10.f };
  int e = t & 15, row = e >> 2, colc = e & 3;
  float v = 0.f;
  if (colc == 3) v = (row < 3) ? tv[row] : 1.f;
  out_sc[(long long)n * 128 + t] = v;
  if (t < 42) out_pos[(long long)n * 42 + t] = 0.f;
}

// ---------------------------------------------------------------------------
// Host orchestration (param order = jax pytree: sorted dict keys, b before w)
// ---------------------------------------------------------------------------
enum {
  I_S = 0, I_Z, I_AATYPE, I_MASK,
  I_AB0L1_B, I_AB0L1_W, I_AB0L2_B, I_AB0L2_W,
  I_AB1L1_B, I_AB1L1_W, I_AB1L2_B, I_AB1L2_W,
  I_ALIN_IN_B, I_ALIN_IN_W, I_ALIN_INIT_B, I_ALIN_INIT_W,
  I_ALIN_OUT_B, I_ALIN_OUT_W,
  I_BB_B, I_BB_W,
  I_IPAB_B, I_IPAB_W, I_HEADW,
  I_KV_B, I_KV_W, I_KVP_B, I_KVP_W,
  I_OUT_B, I_OUT_W, I_Q_B, I_Q_W, I_QP_B, I_QP_W,
  I_LININ_B, I_LININ_W,
  I_LNIPA_B, I_LNIPA_G, I_LNS_B, I_LNS_G, I_LNZ_B, I_LNZ_G,
  I_TL1_B, I_TL1_W, I_TL2_B, I_TL2_W, I_TL3_B, I_TL3_W,
  I_TLN_B, I_TLN_G
};

extern "C" void kernel_launch(void* const* d_in, const int* in_sizes, int n_in,
                              void* d_out, int out_size, void* d_ws, size_t ws_size,
                              hipStream_t stream)
{
  (void)in_sizes; (void)out_size; (void)ws_size;
  const float* s_in = (const float*)d_in[I_S];
  const float* z_in = (const float*)d_in[I_Z];
  const float* mask = (const float*)d_in[I_MASK];
  const float* P[64];
  for (int i = 4; i < n_in && i < 64; ++i) P[i] = (const float*)d_in[i];

  // ---- workspace layout (bump allocator, 256B aligned) ----
  char* wp = (char*)d_ws;
  auto alloc = [&](size_t bytes) { char* p = wp; wp += (bytes + 255) & ~(size_t)255; return p; };
  bf16*  z_bf   = (bf16*) alloc((size_t)NRES*NRES*CZ*2);   // 67 MB, L2-resident
  float* b_pair = (float*)alloc((size_t)NRES*NRES*NH*4);   // pair bias (once)
  float* attn   = (float*)alloc((size_t)NH*NRES*NRES*4);
  float* s_cur  = (float*)alloc((size_t)NRES*CS*4);
  float* s_init = (float*)alloc((size_t)NRES*CS*4);
  float* a0     = (float*)alloc((size_t)NRES*128*4);
  float* qb     = (float*)alloc((size_t)NRES*192*4);
  float* kvb    = (float*)alloc((size_t)NRES*384*4);
  float* qp_raw = (float*)alloc((size_t)NRES*144*4);
  float* qp     = (float*)alloc((size_t)NRES*144*4);
  float* kvp_raw= (float*)alloc((size_t)NRES*432*4);
  float* kvp    = (float*)alloc((size_t)NRES*432*4);
  float* ob     = (float*)alloc((size_t)NRES*192*4);
  float* o_pt   = (float*)alloc((size_t)NRES*96*3*4);
  float* o_pair = (float*)alloc((size_t)NRES*1536*4);
  float* feats  = (float*)alloc((size_t)NRES*2112*4);
  float* t1     = (float*)alloc((size_t)NRES*CS*4);
  float* t2     = (float*)alloc((size_t)NRES*CS*4);
  float* upd    = (float*)alloc((size_t)NRES*16*4);
  float* quat   = (float*)alloc((size_t)NRES*4*4);
  float* trans  = (float*)alloc((size_t)NRES*3*4);
  float* rot    = (float*)alloc((size_t)NRES*9*4);
  float* ar_a   = (float*)alloc((size_t)NRES*128*4);
  float* ar_h   = (float*)alloc((size_t)NRES*128*4);
  float* un_a   = (float*)alloc((size_t)NRES*14*4);

  // ---- d_out layout (concat of tuple outputs, float32) ----
  float* out = (float*)d_out;
  const long long OFF_FRAMES = 0;
  const long long OFF_SC  = OFF_FRAMES + 8LL*NRES*7;
  const long long OFF_UNA = OFF_SC  + 8LL*NRES*128;
  const long long OFF_ANG = OFF_UNA + 8LL*NRES*14;
  const long long OFF_POS = OFF_ANG + 8LL*NRES*14;
  const long long OFF_S   = OFF_POS + 8LL*NRES*42;

  // ================= stage 0: iteration-invariant work =================
  ln_kernel<float><<<NRES, 128, 0, stream>>>(s_in, P[I_LNS_G], P[I_LNS_B], s_init, CS);
  ln_kernel<bf16><<<NRES*NRES, 128, 0, stream>>>(z_in, P[I_LNZ_G], P[I_LNZ_B], z_bf, CZ);

  // s = linear_in(s_init)
  gemm_wmma<4,2,float,float><<<dim3(8,12,1),256,0,stream>>>(
      s_init,0,CS,1,0, P[I_LININ_W],0,CS,1, P[I_LININ_B], nullptr,
      s_cur,0,CS,1, NRES,CS,CS,0);
  // pair bias = ln(z) @ Wb  (262144 x 128 x 12), reads bf16 z
  gemm_wmma<4,2,bf16,float><<<dim3(4096,1,1),256,0,stream>>>(
      z_bf,0,CZ,1,0, P[I_IPAB_W],0,NH,1, P[I_IPAB_B], nullptr,
      b_pair,0,NH,1, NRES*NRES,NH,CZ,0);
  // a0 = lin(relu(s_init), lin_init)
  gemm_wmma<4,2,float,float><<<dim3(8,4,1),256,0,stream>>>(
      s_init,0,CS,1,1, P[I_ALIN_INIT_W],0,128,1, P[I_ALIN_INIT_B], nullptr,
      a0,0,128,1, NRES,128,CS,0);
  init_pose_kernel<<<2,256,0,stream>>>(quat, trans, rot);

  // ================= 8 structure-module blocks =================
  for (int ib = 0; ib < 8; ++ib) {
    // ---- projections ----
    gemm_wmma<4,2,float,float><<<dim3(8,6,1),256,0,stream>>>(
        s_cur,0,CS,1,0, P[I_Q_W],0,192,1, P[I_Q_B], nullptr, qb,0,192,1, NRES,192,CS,0);
    gemm_wmma<4,2,float,float><<<dim3(8,12,1),256,0,stream>>>(
        s_cur,0,CS,1,0, P[I_KV_W],0,384,1, P[I_KV_B], nullptr, kvb,0,384,1, NRES,384,CS,0);
    gemm_wmma<4,2,float,float><<<dim3(8,5,1),256,0,stream>>>(
        s_cur,0,CS,1,0, P[I_QP_W],0,144,1, P[I_QP_B], nullptr, qp_raw,0,144,1, NRES,144,CS,0);
    gemm_wmma<4,2,float,float><<<dim3(8,14,1),256,0,stream>>>(
        s_cur,0,CS,1,0, P[I_KVP_W],0,432,1, P[I_KVP_B], nullptr, kvp_raw,0,432,1, NRES,432,CS,0);
    to_pts_kernel<<<NRES,64,0,stream>>>(qp_raw, rot, trans, qp, 48);
    to_pts_kernel<<<NRES,160,0,stream>>>(kvp_raw, rot, trans, kvp, 144);

    // ---- attention logits: q.k^T per head (K=16, zero-padded to 32) ----
    gemm_wmma<4,2,float,float><<<dim3(8,16,NH),256,0,stream>>>(
        qb,16,192,1,0, kvb,32,1,384, nullptr, nullptr,
        attn,(long long)NRES*NRES,NRES,1, NRES,NRES,CHID,0);
    attn_softmax_kernel<<<dim3(NRES,NH,1),256,0,stream>>>(
        attn, b_pair, qp, kvp, P[I_HEADW], mask);

    // ---- o = a @ v  (per head, N=16) ----
    gemm_wmma<8,1,float,float><<<dim3(4,1,NH),256,0,stream>>>(
        attn,(long long)NRES*NRES,NRES,1,0, kvb+16,32,384,1, nullptr, nullptr,
        ob,16,192,1, NRES,CHID,NRES,0);
    // ---- o_pt = a @ v_pts (per head, N=24) ----
    gemm_wmma<4,2,float,float><<<dim3(8,1,NH),256,0,stream>>>(
        attn,(long long)NRES*NRES,NRES,1,0, kvp+12,36,432,1, nullptr, nullptr,
        o_pt,24,288,1, NRES,PV*3,NRES,0);
    // ---- o_pair = a[:,i,:] @ z_ln[i]  (batched over i, bf16 B from L2) ----
    gemm_wmma<1,8,float,bf16><<<dim3(1,1,NRES),256,0,stream>>>(
        attn,NRES,(long long)NRES*NRES,1,0, z_bf,(long long)NRES*CZ,CZ,1,
        nullptr, nullptr, o_pair,(long long)NH*CZ,CZ,1, NH,CZ,NRES,0);

    // ---- feature concat + output projection (residual into s) ----
    feats_kernel<<<NRES,256,0,stream>>>(ob, o_pt, o_pair, rot, trans, feats);
    gemm_wmma<4,2,float,float><<<dim3(8,12,1),256,0,stream>>>(
        feats,0,2112,1,0, P[I_OUT_W],0,CS,1, P[I_OUT_B], s_cur,
        s_cur,0,CS,1, NRES,CS,2112,0);
    ln_kernel<float><<<NRES,128,0,stream>>>(s_cur, P[I_LNIPA_G], P[I_LNIPA_B], s_cur, CS);

    // ---- transition (1 layer: l1-relu, l2-relu, l3 + residual, ln) ----
    gemm_wmma<4,2,float,float><<<dim3(8,12,1),256,0,stream>>>(
        s_cur,0,CS,1,0, P[I_TL1_W],0,CS,1, P[I_TL1_B], nullptr, t1,0,CS,1, NRES,CS,CS,1);
    gemm_wmma<4,2,float,float><<<dim3(8,12,1),256,0,stream>>>(
        t1,0,CS,1,0, P[I_TL2_W],0,CS,1, P[I_TL2_B], nullptr, t2,0,CS,1, NRES,CS,CS,1);
    gemm_wmma<4,2,float,float><<<dim3(8,12,1),256,0,stream>>>(
        t2,0,CS,1,0, P[I_TL3_W],0,CS,1, P[I_TL3_B], s_cur, s_cur,0,CS,1, NRES,CS,CS,0);
    ln_kernel<float><<<NRES,128,0,stream>>>(s_cur, P[I_TLN_G], P[I_TLN_B], s_cur, CS);

    // ---- backbone update ----
    gemm_wmma<4,2,float,float><<<dim3(8,1,1),256,0,stream>>>(
        s_cur,0,CS,1,0, P[I_BB_W],0,6,1, P[I_BB_B], nullptr, upd,0,16,1, NRES,6,CS,0);
    bb_update_kernel<<<2,256,0,stream>>>(upd, quat, trans, rot,
                                         out + OFF_FRAMES + (long long)ib*NRES*7);

    // ---- angle resnet ----
    gemm_wmma<4,2,float,float><<<dim3(8,4,1),256,0,stream>>>(
        s_cur,0,CS,1,1, P[I_ALIN_IN_W],0,128,1, P[I_ALIN_IN_B], a0,
        ar_a,0,128,1, NRES,128,CS,0);
    const int BW[2][4] = {{I_AB0L1_W,I_AB0L1_B,I_AB0L2_W,I_AB0L2_B},
                          {I_AB1L1_W,I_AB1L1_B,I_AB1L2_W,I_AB1L2_B}};
    for (int blk = 0; blk < 2; ++blk) {
      gemm_wmma<4,2,float,float><<<dim3(8,4,1),256,0,stream>>>(
          ar_a,0,128,1,1, P[BW[blk][0]],0,128,1, P[BW[blk][1]], nullptr,
          ar_h,0,128,1, NRES,128,128,0);
      gemm_wmma<4,2,float,float><<<dim3(8,4,1),256,0,stream>>>(
          ar_h,0,128,1,1, P[BW[blk][2]],0,128,1, P[BW[blk][3]], ar_a,
          ar_a,0,128,1, NRES,128,128,0);
    }
    gemm_wmma<4,2,float,float><<<dim3(8,1,1),256,0,stream>>>(
        ar_a,0,128,1,1, P[I_ALIN_OUT_W],0,14,1, P[I_ALIN_OUT_B], nullptr,
        un_a,0,14,1, NRES,14,128,0);
    angle_out_kernel<<<NRES,32,0,stream>>>(un_a,
        out + OFF_UNA + (long long)ib*NRES*14, out + OFF_ANG + (long long)ib*NRES*14);

    // ---- torsion frames / atom14 (exact under all-zero constant tables) ----
    geom_out_kernel<<<NRES,128,0,stream>>>(trans,
        out + OFF_SC + (long long)ib*NRES*128, out + OFF_POS + (long long)ib*NRES*42);
  }

  // final s output
  hipMemcpyAsync(out + OFF_S, s_cur, (size_t)NRES*CS*4, hipMemcpyDeviceToDevice, stream);
}
